// pattn_9921374453810
// MI455X (gfx1250) — compile-verified, matrix operand order checked by reading, not run on hardware
//
#include <hip/hip_runtime.h>

typedef __attribute__((ext_vector_type(16))) _Float16 v16h;
typedef __attribute__((ext_vector_type(8)))  _Float16 v8h;
typedef __attribute__((ext_vector_type(8)))  float    v8f;
typedef __attribute__((ext_vector_type(4)))  float    v4f;
typedef __attribute__((ext_vector_type(2)))  float    v2f;
typedef __attribute__((ext_vector_type(4)))  unsigned int u32x4;
typedef __attribute__((ext_vector_type(8)))  int      i32x8;
typedef __attribute__((ext_vector_type(4)))  int      i32x4;

constexpr int   HEADS = 12;
constexpr int   SEQ   = 4096;
constexpr int   DH    = 64;
// 1/sqrt(64) * 1/(layer_idx+1=4), with log2(e) folded in: every exp is one v_exp_f32.
constexpr float LOG2E    = 1.44269504088896340736f;
constexpr float QK_SCALE = (1.0f / 32.0f) * LOG2E;

constexpr int BQ   = 64;   // query rows per workgroup (4 waves x 16)
constexpr int KT   = 64;   // key tile: diagonal tile is the ONLY masked tile
constexpr int KPAD = 72;   // K/Vt LDS row stride in halves (144B: 16B aligned, conflict-free)
constexpr int PPAD = 68;   // sP row stride in floats      (272B: 16B aligned, conflict-free)

static __device__ __forceinline__ v16h cat8(v8h a, v8h b) {
  return __builtin_shufflevector(a, b, 0,1,2,3,4,5,6,7,8,9,10,11,12,13,14,15);
}

// ---------------------------------------------------------------------------
// TDM: issue a 2-D f16 tile load (tile_d1 x tile_d0) global -> LDS, padding the
// LDS destination by 16B every 128B so rows land on a 144B (KPAD) pitch.
// D# packing per CDNA5 ISA (group0/group1).
// This toolchain exposes the 6-arg builtin: (g0, g1, g2, g3, g4, cpol).
// ---------------------------------------------------------------------------
static __device__ __forceinline__ void tdm_load_2d_f16(
    unsigned lds_addr, const _Float16* gptr,
    unsigned tensor_d0, unsigned tensor_d1, unsigned d0_stride,
    unsigned tile_d0, unsigned tile_d1)
{
  const unsigned long long ga = (unsigned long long)(uintptr_t)gptr;
  u32x4 g0;
  g0[0] = 1u;                                         // count=1, user mode, no gather
  g0[1] = lds_addr;                                   // lds_addr[31:0]
  g0[2] = (unsigned)(ga & 0xffffffffu);               // global_addr[31:0]
  g0[3] = (unsigned)((ga >> 32) & 0x01ffffffu)        // global_addr[56:32]
        | (2u << 30);                                 // type = 2 ("image")
  i32x8 g1;
  g1[0] = (int)((1u << 16)      // data_size = 1 -> 2 bytes
              | (1u << 20)      // pad_enable
              | (4u << 22)      // pad_interval: 4 -> every 32 DWORDs (128B)
              | (3u << 25));    // pad_amount:   3 -> 4 DWORDs (16B)
  g1[1] = (int)(tensor_d0 << 16);                         // tensor_dim0[15:0]
  g1[2] = (int)((tensor_d0 >> 16) | (tensor_d1 << 16));   // dim0 hi | dim1 lo
  g1[3] = (int)((tensor_d1 >> 16) | (tile_d0 << 16));     // dim1 hi | tile_dim0
  g1[4] = (int)(tile_d1 & 0xffffu);                       // tile_dim1 (tile_dim2=0)
  g1[5] = (int)d0_stride;                                 // tensor_dim0_stride[31:0]
  g1[6] = 0;                                              // stride hi, dim1_stride lo
  g1[7] = 0;
  const i32x4 z4 = {0, 0, 0, 0};
  const i32x8 z8 = {0, 0, 0, 0, 0, 0, 0, 0};
  __builtin_amdgcn_tensor_load_to_lds(g0, g1, z4, z4, z8, 0);
}

// ---------------------------------------------------------------------------
// Pre-pass: f32 -> f16. K kept [S][D]; V transposed to [D][S] per head so the
// attention kernel's TDM loads are dense 2-D tiles in both cases.
// ---------------------------------------------------------------------------
__global__ __launch_bounds__(256) void prep_kernel(const float* __restrict__ k,
                                                   const float* __restrict__ v,
                                                   _Float16* __restrict__ kf,
                                                   _Float16* __restrict__ vt) {
  __shared__ _Float16 tile[64][KPAD];
  const int h   = blockIdx.y;
  const int s0  = blockIdx.x * 64;
  const int tid = threadIdx.x;
  const float*    Kp = k  + (size_t)h * SEQ * DH;
  const float*    Vp = v  + (size_t)h * SEQ * DH;
  _Float16*       Kf = kf + (size_t)h * SEQ * DH;
  _Float16*       Vt = vt + (size_t)h * DH * SEQ;

#pragma unroll
  for (int i = 0; i < 8; ++i) {          // K: straight convert, 64x64 tile
    int e = (tid + i * 256) * 2;
    int s = e >> 6, d = e & 63;
    v2f x = *(const v2f*)(Kp + (size_t)(s0 + s) * DH + d);
    Kf[(size_t)(s0 + s) * DH + d]     = (_Float16)x.x;
    Kf[(size_t)(s0 + s) * DH + d + 1] = (_Float16)x.y;
  }
#pragma unroll
  for (int i = 0; i < 8; ++i) {          // V: stage tile in LDS
    int e = (tid + i * 256) * 2;
    int s = e >> 6, d = e & 63;
    v2f x = *(const v2f*)(Vp + (size_t)(s0 + s) * DH + d);
    tile[s][d]     = (_Float16)x.x;
    tile[s][d + 1] = (_Float16)x.y;
  }
  __syncthreads();
#pragma unroll
  for (int i = 0; i < 8; ++i) {          // V: write transposed, coalesced
    int e = (tid + i * 256) * 2;
    int d = e >> 6, s = e & 63;
    Vt[(size_t)d * SEQ + s0 + s]     = tile[s][d];
    Vt[(size_t)d * SEQ + s0 + s + 1] = tile[s + 1][d];
  }
}

// ---------------------------------------------------------------------------
// Flash attention, TDM-fed double-buffered K/V tiles.
// ---------------------------------------------------------------------------
__global__ __launch_bounds__(128) void fattn_tdm_kernel(const float* __restrict__ q,
                                                        const _Float16* __restrict__ kf,
                                                        const _Float16* __restrict__ vt,
                                                        float* __restrict__ out) {
  __shared__ _Float16 sK [2][KT][KPAD];   // K tile, [key][dim] f16
  __shared__ _Float16 sVt[2][DH][KPAD];   // V tile, [dim][key] f16
  __shared__ float    sP [4][16][PPAD];   // per-wave P re-layout buffer

  const int h   = blockIdx.y;
  const int qb  = blockIdx.x * BQ;
  const int tid = threadIdx.x;
  const int wv  = tid >> 5;
  const int ln  = tid & 31;
  const int lh  = ln & 15;
  const int hi  = ln >> 4;

  const float*    Q  = q  + (size_t)h * SEQ * DH;
  const _Float16* Kf = kf + (size_t)h * SEQ * DH;
  const _Float16* Vt = vt + (size_t)h * DH * SEQ;
  float*          O  = out + (size_t)h * SEQ * DH;

  const int wqb  = qb + wv * 16;
  const int qrow = wqb + lh;

  // Q A-fragments (scale*log2e folded in)
  v16h aq[2];
#pragma unroll
  for (int c = 0; c < 2; ++c) {
    const float* src = Q + (size_t)qrow * DH + c * 32 + hi * 8;
    v4f x0 = *(const v4f*)(src);
    v4f x1 = *(const v4f*)(src + 4);
    v4f x2 = *(const v4f*)(src + 16);
    v4f x3 = *(const v4f*)(src + 20);
#pragma unroll
    for (int i = 0; i < 4; ++i) {
      aq[c][i]      = (_Float16)(x0[i] * QK_SCALE);
      aq[c][4 + i]  = (_Float16)(x1[i] * QK_SCALE);
      aq[c][8 + i]  = (_Float16)(x2[i] * QK_SCALE);
      aq[c][12 + i] = (_Float16)(x3[i] * QK_SCALE);
    }
  }

  v16h vones;
#pragma unroll
  for (int i = 0; i < 16; ++i) vones[i] = (_Float16)1.0f;

  const v8f vzero = {0.f, 0.f, 0.f, 0.f, 0.f, 0.f, 0.f, 0.f};
  v8f oacc[4] = {vzero, vzero, vzero, vzero};
  float m_i[8], l_i[8];
#pragma unroll
  for (int r = 0; r < 8; ++r) { m_i[r] = -__builtin_inff(); l_i[r] = 0.0f; }

  const int nkt = (qb + BQ) / KT;

  // prologue: wave 0 requests tile 0 into buffer 0
  if (wv == 0) {
    tdm_load_2d_f16((unsigned)(uintptr_t)&sK[0][0][0],  Kf + (size_t)0 * DH,
                    DH, SEQ, DH, DH, KT);
    tdm_load_2d_f16((unsigned)(uintptr_t)&sVt[0][0][0], Vt + 0,
                    SEQ, DH, SEQ, KT, DH);
  }

  for (int t = 0; t < nkt; ++t) {
    const int kt = t * KT;
    const int b  = t & 1;
    __syncthreads();  // all waves done reading buffer b^1 (tile t-1)

    if (wv == 0) {
      if (t + 1 < nkt) {        // prefetch tile t+1 into the other buffer
        const int kn = (t + 1) * KT;
        tdm_load_2d_f16((unsigned)(uintptr_t)&sK[b ^ 1][0][0],  Kf + (size_t)kn * DH,
                        DH, SEQ, DH, DH, KT);
        tdm_load_2d_f16((unsigned)(uintptr_t)&sVt[b ^ 1][0][0], Vt + kn,
                        SEQ, DH, SEQ, KT, DH);
        __builtin_amdgcn_s_wait_tensorcnt(2);  // tile t's 2 in-order ops done
      } else {
        __builtin_amdgcn_s_wait_tensorcnt(0);
      }
    }
    __syncthreads();  // tile t resident in buffer b for everyone

    // ---- S = (Q*scale*log2e) @ K^T ----
    v8f sc[4];
#pragma unroll
    for (int n = 0; n < 4; ++n) {
      v8f acc = vzero;
#pragma unroll
      for (int c = 0; c < 2; ++c) {
        const _Float16* p = &sK[b][n * 16 + lh][c * 32 + hi * 8];
        v16h bk = cat8(*(const v8h*)p, *(const v8h*)(p + 16));
        acc = __builtin_amdgcn_wmma_f32_16x16x32_f16(false, aq[c], false, bk,
                                                     (short)0, acc, false, false);
      }
      sc[n] = acc;
    }

    // ---- causal mask: only the diagonal tile ----
    if (kt + KT - 1 > wqb) {
#pragma unroll
      for (int r = 0; r < 8; ++r) {
        const int qr = wqb + r + hi * 8;
#pragma unroll
        for (int n = 0; n < 4; ++n) {
          const int ky = kt + n * 16 + lh;
          if (ky > qr) sc[n][r] = -__builtin_inff();
        }
      }
    }

    // ---- online softmax ----
    float corr[8];
#pragma unroll
    for (int r = 0; r < 8; ++r) {
      float t0 = fmaxf(fmaxf(sc[0][r], sc[1][r]), fmaxf(sc[2][r], sc[3][r]));
      t0 = fmaxf(t0, __shfl_xor(t0, 1));
      t0 = fmaxf(t0, __shfl_xor(t0, 2));
      t0 = fmaxf(t0, __shfl_xor(t0, 4));
      t0 = fmaxf(t0, __shfl_xor(t0, 8));
      const float mnew = fmaxf(m_i[r], t0);
      corr[r] = __builtin_amdgcn_exp2f(m_i[r] - mnew);
      m_i[r]  = mnew;
    }
#pragma unroll
    for (int n = 0; n < 4; ++n)
#pragma unroll
      for (int r = 0; r < 8; ++r)
        sc[n][r] = __builtin_amdgcn_exp2f(sc[n][r] - m_i[r]);
#pragma unroll
    for (int d = 0; d < 4; ++d)
#pragma unroll
      for (int r = 0; r < 8; ++r) oacc[d][r] *= corr[r];

    // ---- P: C layout -> A layout via per-wave LDS ----
#pragma unroll
    for (int n = 0; n < 4; ++n)
#pragma unroll
      for (int r = 0; r < 8; ++r)
        sP[wv][r + hi * 8][n * 16 + lh] = sc[n][r];

    v16h ap[2];
#pragma unroll
    for (int c = 0; c < 2; ++c) {
      const float* pp = &sP[wv][lh][c * 32 + hi * 8];
#pragma unroll
      for (int i = 0; i < 8; ++i) {
        ap[c][i]     = (_Float16)pp[i];
        ap[c][8 + i] = (_Float16)pp[16 + i];
      }
    }

    // ---- row sum on the WMMA pipe ----
    {
      v8f ls = vzero;
      ls = __builtin_amdgcn_wmma_f32_16x16x32_f16(false, ap[0], false, vones,
                                                  (short)0, ls, false, false);
      ls = __builtin_amdgcn_wmma_f32_16x16x32_f16(false, ap[1], false, vones,
                                                  (short)0, ls, false, false);
#pragma unroll
      for (int r = 0; r < 8; ++r) l_i[r] = l_i[r] * corr[r] + ls[r];
    }

    // ---- O += P @ V ----
#pragma unroll
    for (int d = 0; d < 4; ++d) {
#pragma unroll
      for (int c = 0; c < 2; ++c) {
        const _Float16* pv = &sVt[b][d * 16 + lh][c * 32 + hi * 8];
        v16h bv = cat8(*(const v8h*)pv, *(const v8h*)(pv + 16));
        oacc[d] = __builtin_amdgcn_wmma_f32_16x16x32_f16(false, ap[c], false, bv,
                                                         (short)0, oacc[d], false, false);
      }
    }
  }

#pragma unroll
  for (int r = 0; r < 8; ++r) {
    const float inv = 1.0f / l_i[r];
    const int   qr  = wqb + r + hi * 8;
#pragma unroll
    for (int d = 0; d < 4; ++d)
      O[(size_t)qr * DH + d * 16 + lh] = oacc[d][r] * inv;
  }
}

// ---------------------------------------------------------------------------
// Fallback (self-contained, no workspace): round-2 kernel.
// ---------------------------------------------------------------------------
__global__ __launch_bounds__(128) void fattn_fb_kernel(const float* __restrict__ q,
                                                       const float* __restrict__ k,
                                                       const float* __restrict__ v,
                                                       float* __restrict__ out) {
  __shared__ _Float16 sK [KT][KPAD];
  __shared__ _Float16 sVt[DH][KPAD];
  __shared__ float    sP [4][16][PPAD];

  const int h   = blockIdx.y;
  const int qb  = blockIdx.x * BQ;
  const int tid = threadIdx.x;
  const int wv  = tid >> 5;
  const int ln  = tid & 31;
  const int lh  = ln & 15;
  const int hi  = ln >> 4;

  const size_t hoff = (size_t)h * SEQ * DH;
  const float* Q  = q + hoff;
  const float* Kp = k + hoff;
  const float* Vp = v + hoff;
  float*       O  = out + hoff;

  const int wqb  = qb + wv * 16;
  const int qrow = wqb + lh;

  v16h aq[2];
#pragma unroll
  for (int c = 0; c < 2; ++c) {
    const float* src = Q + (size_t)qrow * DH + c * 32 + hi * 8;
    v4f x0 = *(const v4f*)(src);
    v4f x1 = *(const v4f*)(src + 4);
    v4f x2 = *(const v4f*)(src + 16);
    v4f x3 = *(const v4f*)(src + 20);
#pragma unroll
    for (int i = 0; i < 4; ++i) {
      aq[c][i]      = (_Float16)(x0[i] * QK_SCALE);
      aq[c][4 + i]  = (_Float16)(x1[i] * QK_SCALE);
      aq[c][8 + i]  = (_Float16)(x2[i] * QK_SCALE);
      aq[c][12 + i] = (_Float16)(x3[i] * QK_SCALE);
    }
  }

  v16h vones;
#pragma unroll
  for (int i = 0; i < 16; ++i) vones[i] = (_Float16)1.0f;

  const v8f vzero = {0.f, 0.f, 0.f, 0.f, 0.f, 0.f, 0.f, 0.f};
  v8f oacc[4] = {vzero, vzero, vzero, vzero};
  float m_i[8], l_i[8];
#pragma unroll
  for (int r = 0; r < 8; ++r) { m_i[r] = -__builtin_inff(); l_i[r] = 0.0f; }

  const int nkt = (qb + BQ) / KT;
  for (int t = 0; t < nkt; ++t) {
    const int kt = t * KT;
    __syncthreads();
#pragma unroll
    for (int i = 0; i < 16; ++i) {
      int e  = (tid + i * 128) * 2;
      int ky = e >> 6, dm = e & 63;
      v2f kv = *(const v2f*)(Kp + (size_t)(kt + ky) * DH + dm);
      sK[ky][dm]     = (_Float16)kv.x;
      sK[ky][dm + 1] = (_Float16)kv.y;
    }
#pragma unroll
    for (int i = 0; i < 16; ++i) {
      int u  = tid + i * 128;
      int dm = u & 63, k2 = (u >> 6) * 2;
      float a  = Vp[(size_t)(kt + k2) * DH + dm];
      float bb = Vp[(size_t)(kt + k2 + 1) * DH + dm];
      sVt[dm][k2]     = (_Float16)a;
      sVt[dm][k2 + 1] = (_Float16)bb;
    }
    __syncthreads();

    v8f sc[4];
#pragma unroll
    for (int n = 0; n < 4; ++n) {
      v8f acc = vzero;
#pragma unroll
      for (int c = 0; c < 2; ++c) {
        const _Float16* p = &sK[n * 16 + lh][c * 32 + hi * 8];
        v16h bk = cat8(*(const v8h*)p, *(const v8h*)(p + 16));
        acc = __builtin_amdgcn_wmma_f32_16x16x32_f16(false, aq[c], false, bk,
                                                     (short)0, acc, false, false);
      }
      sc[n] = acc;
    }

    if (kt + KT - 1 > wqb) {
#pragma unroll
      for (int r = 0; r < 8; ++r) {
        const int qr = wqb + r + hi * 8;
#pragma unroll
        for (int n = 0; n < 4; ++n) {
          const int ky = kt + n * 16 + lh;
          if (ky > qr) sc[n][r] = -__builtin_inff();
        }
      }
    }

    float corr[8];
#pragma unroll
    for (int r = 0; r < 8; ++r) {
      float t0 = fmaxf(fmaxf(sc[0][r], sc[1][r]), fmaxf(sc[2][r], sc[3][r]));
      t0 = fmaxf(t0, __shfl_xor(t0, 1));
      t0 = fmaxf(t0, __shfl_xor(t0, 2));
      t0 = fmaxf(t0, __shfl_xor(t0, 4));
      t0 = fmaxf(t0, __shfl_xor(t0, 8));
      const float mnew = fmaxf(m_i[r], t0);
      corr[r] = __builtin_amdgcn_exp2f(m_i[r] - mnew);
      m_i[r]  = mnew;
    }
#pragma unroll
    for (int n = 0; n < 4; ++n)
#pragma unroll
      for (int r = 0; r < 8; ++r)
        sc[n][r] = __builtin_amdgcn_exp2f(sc[n][r] - m_i[r]);
#pragma unroll
    for (int d = 0; d < 4; ++d)
#pragma unroll
      for (int r = 0; r < 8; ++r) oacc[d][r] *= corr[r];

#pragma unroll
    for (int n = 0; n < 4; ++n)
#pragma unroll
      for (int r = 0; r < 8; ++r)
        sP[wv][r + hi * 8][n * 16 + lh] = sc[n][r];

    v16h ap[2];
#pragma unroll
    for (int c = 0; c < 2; ++c) {
      const float* pp = &sP[wv][lh][c * 32 + hi * 8];
#pragma unroll
      for (int i = 0; i < 8; ++i) {
        ap[c][i]     = (_Float16)pp[i];
        ap[c][8 + i] = (_Float16)pp[16 + i];
      }
    }

    {
      v8f ls = vzero;
      ls = __builtin_amdgcn_wmma_f32_16x16x32_f16(false, ap[0], false, vones,
                                                  (short)0, ls, false, false);
      ls = __builtin_amdgcn_wmma_f32_16x16x32_f16(false, ap[1], false, vones,
                                                  (short)0, ls, false, false);
#pragma unroll
      for (int r = 0; r < 8; ++r) l_i[r] = l_i[r] * corr[r] + ls[r];
    }

#pragma unroll
    for (int d = 0; d < 4; ++d) {
#pragma unroll
      for (int c = 0; c < 2; ++c) {
        const _Float16* pv = &sVt[d * 16 + lh][c * 32 + hi * 8];
        v16h bv = cat8(*(const v8h*)pv, *(const v8h*)(pv + 16));
        oacc[d] = __builtin_amdgcn_wmma_f32_16x16x32_f16(false, ap[c], false, bv,
                                                         (short)0, oacc[d], false, false);
      }
    }
  }

#pragma unroll
  for (int r = 0; r < 8; ++r) {
    const float inv = 1.0f / l_i[r];
    const int   qr  = wqb + r + hi * 8;
#pragma unroll
    for (int d = 0; d < 4; ++d)
      O[(size_t)qr * DH + d * 16 + lh] = oacc[d][r] * inv;
  }
}

extern "C" void kernel_launch(void* const* d_in, const int* in_sizes, int n_in,
                              void* d_out, int out_size, void* d_ws, size_t ws_size,
                              hipStream_t stream) {
  const float* q = (const float*)d_in[0];
  const float* k = (const float*)d_in[1];
  const float* v = (const float*)d_in[2];
  float*       o = (float*)d_out;

  const size_t halfBytes = (size_t)HEADS * SEQ * DH * 2;  // 6,291,456
  dim3 grid(SEQ / BQ, HEADS, 1);

  if (ws_size >= 2 * halfBytes) {
    _Float16* kf = (_Float16*)d_ws;
    _Float16* vt = (_Float16*)((char*)d_ws + halfBytes);
    prep_kernel<<<dim3(SEQ / 64, HEADS, 1), dim3(256, 1, 1), 0, stream>>>(k, v, kf, vt);
    fattn_tdm_kernel<<<grid, dim3(128, 1, 1), 0, stream>>>(q, kf, vt, o);
  } else {
    fattn_fb_kernel<<<grid, dim3(128, 1, 1), 0, stream>>>(q, k, v, o);
  }
}
